// FAM1_35510789604031
// MI455X (gfx1250) — matrix operand order repaired and checked by reading, not compiled, not run on hardware
//
#include <hip/hip_runtime.h>
#include <hip/hip_bf16.h>

// ---------------- problem constants ----------------
#define DGN 8
#define KKN 9
#define Bn  2
#define C1  256
#define C2  128
#define Hn  160
#define Wn  160
#define Nn  (Hn * Wn)      // 25600
#define MOM 216            // 3*DG*KK
#define MOMP 224           // padded to 14*16
#define CGN 16             // c2 / DG
#define KDCN 160           // per-group K (9*16=144) padded to 5*32

typedef __attribute__((ext_vector_type(16))) __bf16 v16bf;
typedef __attribute__((ext_vector_type(8)))  float  v8f;
typedef unsigned short u16;
typedef unsigned int   u32;

// ---------------- helpers ----------------
static __device__ __forceinline__ u16 f2bf(float f) {
    u32 u = __float_as_uint(f);
    u32 r = u + 0x7FFFu + ((u >> 16) & 1u);   // round-to-nearest-even
    return (u16)(r >> 16);
}
static __device__ __forceinline__ float sigf(float x) {
    return 1.0f / (1.0f + __expf(-x));
}
// two 16B chunks -> one 16-element bf16 fragment
static __device__ __forceinline__ v16bf frag_ld(const u16* p0, const u16* p1) {
    union { v16bf v; uint4 q[2]; } f;
    f.q[0] = *reinterpret_cast<const uint4*>(p0);
    f.q[1] = *reinterpret_cast<const uint4*>(p1);
    return f.v;
}
static __device__ __forceinline__ v8f wmma_bf16(v16bf a, v16bf b, v8f c) {
    return __builtin_amdgcn_wmma_f32_16x16x32_bf16(false, a, false, b, (short)0, c, false, false);
}
static __device__ __forceinline__ v8f zero8() {
    v8f z = {0.f, 0.f, 0.f, 0.f, 0.f, 0.f, 0.f, 0.f};
    return z;
}
// async global->LDS copy of 16 bytes (ASYNCcnt-tracked, CDNA5)
static __device__ __forceinline__ void async_b128(u32 lds_off, u32 goff,
                                                  unsigned long long gbase) {
    asm volatile("global_load_async_to_lds_b128 %0, %1, %2"
                 :: "v"(lds_off), "v"(goff), "s"(gbase)
                 : "memory");
}
static __device__ __forceinline__ void wait_async0() {
    asm volatile("s_wait_asynccnt 0x0" ::: "memory");
}

// ---------------- 1: global average pool ----------------
__global__ void k_pool(const float* __restrict__ fl, float* __restrict__ g) {
    int bc = blockIdx.x;                      // b*C1 + c
    const float* p = fl + (size_t)bc * Nn;
    float s = 0.f;
    for (int i = threadIdx.x; i < Nn; i += 256) s += p[i];
    __shared__ float red[256];
    red[threadIdx.x] = s;
    __syncthreads();
    for (int st = 128; st > 0; st >>= 1) {
        if (threadIdx.x < st) red[threadIdx.x] += red[threadIdx.x + st];
        __syncthreads();
    }
    if (threadIdx.x == 0) g[bc] = red[0] * (1.0f / Nn);
}

// ---------------- 2: attention -> per-channel scale (1 + sigmoid) --------
__global__ void k_atten(const float* __restrict__ g, const float* __restrict__ wat,
                        float* __restrict__ scale) {
    int b = blockIdx.x;
    __shared__ float gs[C1];
    gs[threadIdx.x] = g[b * C1 + threadIdx.x];
    __syncthreads();
    int c = threadIdx.x;
    const float* wr = wat + (size_t)c * C1;
    float s = 0.f;
    for (int j = 0; j < C1; j++) s += wr[j] * gs[j];
    scale[b * C1 + c] = 1.0f + sigf(s);
}

// ---------------- 3: weight prep (fp32 -> bf16, WMMA layouts) ------------
__global__ void k_prep_w(const float* __restrict__ w_conv, const float* __restrict__ w_offw,
                         const float* __restrict__ w_omw,  const float* __restrict__ w_dcn,
                         u16* __restrict__ w_arm, u16* __restrict__ w_off,
                         u16* __restrict__ w_om,  u16* __restrict__ w_dA) {
    int i = blockIdx.x * 256 + threadIdx.x;
    if (i < C2 * C1) { w_arm[i] = f2bf(w_conv[i]); return; }
    i -= C2 * C1;
    if (i < C2 * C1) { w_off[i] = f2bf(w_offw[i]); return; }
    i -= C2 * C1;
    if (i < 9 * MOMP * C2) {              // [ky*3+kx][o(pad224)][c]
        int kyx = i / (MOMP * C2);
        int r   = i - kyx * MOMP * C2;
        int o   = r / C2;
        int c   = r - o * C2;
        float v = (o < MOM) ? w_omw[((size_t)(o * C2 + c)) * 9 + kyx] : 0.f;
        w_om[i] = f2bf(v);
        return;
    }
    i -= 9 * MOMP * C2;
    if (i < DGN * C2 * KDCN) {            // [d][o][k*16+cg (pad160)]
        int d  = i / (C2 * KDCN);
        int r  = i - d * C2 * KDCN;
        int o  = r / KDCN;
        int kc = r - o * KDCN;
        float v = 0.f;
        if (kc < KKN * CGN) {
            int k = kc >> 4, cg = kc & 15;
            v = w_dcn[((size_t)(o * C2 + d * CGN + cg)) * KKN + k];
        }
        w_dA[i] = f2bf(v);
    }
}

// ---------------- 4: feat_l*scale -> NHWC bf16 (tiled transpose) ---------
__global__ void k_trans_l(const float* __restrict__ fl, const float* __restrict__ scale,
                          u16* __restrict__ xl) {
    int b = blockIdx.z, c0 = blockIdx.y * 64, n0 = blockIdx.x * 64;
    __shared__ float t[64][65];
    int col = threadIdx.x & 63, rb = threadIdx.x >> 6;
#pragma unroll
    for (int i = 0; i < 16; i++) {
        int row = rb + i * 4;                       // channel local
        t[row][col] = fl[((size_t)(b * C1 + c0 + row)) * Nn + n0 + col] *
                      scale[b * C1 + c0 + row];
    }
    __syncthreads();
#pragma unroll
    for (int i = 0; i < 16; i++) {
        int nrow = rb + i * 4;                      // pixel local
        xl[((size_t)b * Nn + n0 + nrow) * C1 + c0 + col] = f2bf(t[col][nrow]);
    }
}

// ---------------- 5: feat_s -> (2x into concat NHWC) + grouped bf16 ------
__global__ void k_trans_s(const float* __restrict__ fs, u16* __restrict__ xarm2,
                          u16* __restrict__ xsg) {
    int b = blockIdx.z, c0 = blockIdx.y * 64, n0 = blockIdx.x * 64;
    __shared__ float t[64][65];
    int col = threadIdx.x & 63, rb = threadIdx.x >> 6;
#pragma unroll
    for (int i = 0; i < 16; i++) {
        int row = rb + i * 4;
        t[row][col] = fs[((size_t)(b * C2 + c0 + row)) * Nn + n0 + col];
    }
    __syncthreads();
#pragma unroll
    for (int i = 0; i < 16; i++) {
        int nrow = rb + i * 4;
        float v = t[col][nrow];
        int c = c0 + col, n = n0 + nrow;
        xarm2[((size_t)b * Nn + n) * C1 + C2 + c] = f2bf(2.0f * v);   // offset path
        int d = c >> 4, cg = c & 15;
        xsg[(((size_t)(b * DGN + d)) * Nn + n) * CGN + cg] = f2bf(v); // gather path
    }
}

// ---------------- 6/7: 1x1-conv GEMM, M=128, K=256 -----------------------
// B tile (64 pixels x K=256 bf16) staged once per block via async
// global->LDS copies (ASYNCcnt), then consumed 8x by the waves from LDS.
// MODE 0: C -> feat_arm fp32 NCHW + bf16 into xarm2[:,0:128]
// MODE 1: C -> off_nhwc bf16
#define BPITCH 264      // LDS row pitch in halfwords (256 + 8 pad)
template <int MODE>
__global__ void k_gemm1x1(const u16* __restrict__ wA, const u16* __restrict__ xB,
                          float* __restrict__ feat_arm, u16* __restrict__ outB) {
    int b = blockIdx.y, n0 = blockIdx.x * 64;
    int lane = threadIdx.x & 31, wid = threadIdx.x >> 5;
    int h = lane >> 4, lr = lane & 15;
    int m0 = wid * 16;

    __shared__ u16 sX[64 * BPITCH];     // [pixel][K], pitch-padded
    // ---- async stage: 64 rows x 512B, 2048 x b128 chunks ----
    {
        u32 ldsbase = (u32)(uintptr_t)&sX[0];
        unsigned long long gbase =
            (unsigned long long)(uintptr_t)(xB + ((size_t)b * Nn + n0) * C1);
        for (int i = threadIdx.x; i < 64 * 32; i += 256) {
            int row = i >> 5, qc = i & 31;           // qc: 16B chunk within row
            u32 lds_off = ldsbase + (u32)(row * BPITCH * 2 + qc * 16);
            u32 goff    = (u32)(row * (C1 * 2) + qc * 16);
            async_b128(lds_off, goff, gbase);
        }
        wait_async0();
    }
    __syncthreads();

    v8f acc[4];
#pragma unroll
    for (int i = 0; i < 4; i++) acc[i] = zero8();

    const u16* arow = wA + (size_t)(m0 + lr) * C1;
    for (int kc = 0; kc < 8; kc++) {
        v16bf a = frag_ld(arow + kc * 32 + h * 8, arow + kc * 32 + 16 + h * 8);
#pragma unroll
        for (int nf = 0; nf < 4; nf++) {
            const u16* brow = &sX[(nf * 16 + lr) * BPITCH + kc * 32 + h * 8];
            v16bf bb = frag_ld(brow, brow + 16);
            acc[nf] = wmma_bf16(a, bb, acc[nf]);
        }
    }
#pragma unroll
    for (int nf = 0; nf < 4; nf++)
#pragma unroll
        for (int r = 0; r < 8; r++) {
            int m = m0 + r + 8 * h;
            int n = n0 + nf * 16 + lr;
            float v = acc[nf][r];
            if (MODE == 0) {
                feat_arm[((size_t)(b * C2 + m)) * Nn + n] = v;
                outB[((size_t)b * Nn + n) * C1 + m] = f2bf(v);
            } else {
                outB[((size_t)b * Nn + n) * C2 + m] = f2bf(v);
            }
        }
}

// ---------------- 8: 3x3 conv for offsets/mask (implicit GEMM) -----------
__global__ void k_conv_om(const u16* __restrict__ w_om, const u16* __restrict__ offh,
                          const float* __restrict__ bias, float* __restrict__ om) {
    int y = blockIdx.x, mblk = blockIdx.y, b = blockIdx.z;
    int lane = threadIdx.x & 31, wid = threadIdx.x >> 5;
    int h = lane >> 4, lr = lane & 15;
    int m0 = mblk * 128 + wid * 16;

    __shared__ u16 sB[162 * 136];   // [x_in + 1][c], halo cols, pitch 136
    v8f acc[10];
#pragma unroll
    for (int i = 0; i < 10; i++) acc[i] = zero8();

    for (int ky = 0; ky < 3; ky++) {
        int yy = y + ky - 1;
        __syncthreads();
        for (int idx = threadIdx.x; idx < 162 * 16; idx += 256) {   // 16 uint4 per row
            int row = idx >> 4, qc = idx & 15, xin = row - 1;
            uint4 v = make_uint4(0u, 0u, 0u, 0u);
            if (yy >= 0 && yy < Hn && xin >= 0 && xin < Wn)
                v = *reinterpret_cast<const uint4*>(
                        offh + ((size_t)(b * Nn + yy * Wn + xin)) * C2 + qc * 8);
            *reinterpret_cast<uint4*>(&sB[row * 136 + qc * 8]) = v;
        }
        __syncthreads();
        for (int kx = 0; kx < 3; kx++) {
            const u16* arow = w_om + ((size_t)((ky * 3 + kx) * MOMP + m0 + lr)) * C2;
            for (int kc = 0; kc < 4; kc++) {
                v16bf a = frag_ld(arow + kc * 32 + h * 8, arow + kc * 32 + 16 + h * 8);
#pragma unroll
                for (int nf = 0; nf < 10; nf++) {
                    const u16* brow = &sB[(nf * 16 + lr + kx) * 136 + kc * 32 + h * 8];
                    v16bf bb = frag_ld(brow, brow + 16);
                    acc[nf] = wmma_bf16(a, bb, acc[nf]);
                }
            }
        }
    }
#pragma unroll
    for (int r = 0; r < 8; r++) {
        int m = m0 + r + 8 * h;
        if (m < MOM) {
            float bs = bias[m];
#pragma unroll
            for (int nf = 0; nf < 10; nf++) {
                int x = nf * 16 + lr;
                om[((size_t)(b * MOM + m)) * Nn + y * Wn + x] = acc[nf][r] + bs;
            }
        }
    }
}

// ---------------- 9: fused bilinear gather + DCN GEMM + epilogue ---------
__global__ void k_dcn(const u16* __restrict__ wdA, const u16* __restrict__ xsg,
                      const float* __restrict__ om, const float* __restrict__ dcnb,
                      const float* __restrict__ feat_arm, float* __restrict__ out) {
    int b = blockIdx.y, n0 = blockIdx.x * 64;
    int lane = threadIdx.x & 31, wid = threadIdx.x >> 5;
    int h = lane >> 4, lr = lane & 15;
    int m0 = wid * 16;

    __shared__ u16 sV[64 * 168];    // [nloc][K 0..159], pitch 168 halfwords
    for (int i = threadIdx.x; i < 64 * 168 / 8; i += 256)
        *reinterpret_cast<uint4*>(&sV[i * 8]) = make_uint4(0u, 0u, 0u, 0u);

    v8f acc[4];
#pragma unroll
    for (int i = 0; i < 4; i++) acc[i] = zero8();

    for (int dg = 0; dg < DGN; dg++) {
        __syncthreads();    // previous WMMA reads finished before overwrite
        // ---- gather phase: val tile [K=144][64] bf16 into LDS ----
        for (int task = threadIdx.x; task < KKN * 64; task += 256) {
            int nloc = task & 63, k = task >> 6;
            int n = n0 + nloc;
            int yq = n / Wn, xq = n - yq * Wn;
            size_t ob = ((size_t)(b * MOM) + dg * KKN + k) * Nn + n;
            float dy = om[ob];
            float dx = om[ob + (size_t)(DGN * KKN) * Nn];
            float mk = sigf(om[ob + (size_t)(2 * DGN * KKN) * Nn]);
            float py = (float)yq + (float)(k / 3 - 1) + dy;
            float px = (float)xq + (float)(k % 3 - 1) + dx;
            float y0f = floorf(py), x0f = floorf(px);
            float wy = py - y0f, wx = px - x0f;
            int y0 = (int)y0f, x0 = (int)x0f;

            float a[16];
#pragma unroll
            for (int j = 0; j < 16; j++) a[j] = 0.f;
            const u16* base = xsg + ((size_t)(b * DGN + dg)) * Nn * CGN;
#pragma unroll
            for (int cy = 0; cy < 2; cy++) {
                int yi = y0 + cy;
                if (yi < 0 || yi >= Hn) continue;
                float wyc = cy ? wy : (1.0f - wy);
#pragma unroll
                for (int cx = 0; cx < 2; cx++) {
                    int xi = x0 + cx;
                    if (xi < 0 || xi >= Wn) continue;
                    float wgt = wyc * (cx ? wx : (1.0f - wx));
                    const uint4* p = reinterpret_cast<const uint4*>(
                        base + ((size_t)(yi * Wn + xi)) * CGN);
                    uint4 q0 = p[0], q1 = p[1];
                    u32 qs[8] = {q0.x, q0.y, q0.z, q0.w, q1.x, q1.y, q1.z, q1.w};
#pragma unroll
                    for (int j = 0; j < 8; j++) {
                        a[2 * j]     += wgt * __uint_as_float(qs[j] << 16);
                        a[2 * j + 1] += wgt * __uint_as_float(qs[j] & 0xFFFF0000u);
                    }
                }
            }
            u32 pk[8];
#pragma unroll
            for (int j = 0; j < 8; j++) {
                u32 lo = f2bf(a[2 * j] * mk);
                u32 hi = f2bf(a[2 * j + 1] * mk);
                pk[j] = lo | (hi << 16);
            }
            uint4* dst = reinterpret_cast<uint4*>(&sV[nloc * 168 + k * 16]);
            dst[0] = make_uint4(pk[0], pk[1], pk[2], pk[3]);
            dst[1] = make_uint4(pk[4], pk[5], pk[6], pk[7]);
        }
        __syncthreads();
        // ---- WMMA phase: 5 K-steps (K=160, zero padded) x 4 N-frags ----
        const u16* arow = wdA + ((size_t)(dg * C2 + m0 + lr)) * KDCN;
        for (int kc = 0; kc < 5; kc++) {
            v16bf a = frag_ld(arow + kc * 32 + h * 8, arow + kc * 32 + 16 + h * 8);
#pragma unroll
            for (int nf = 0; nf < 4; nf++) {
                const u16* brow = &sV[(nf * 16 + lr) * 168 + kc * 32 + h * 8];
                v16bf bb = frag_ld(brow, brow + 16);
                acc[nf] = wmma_bf16(a, bb, acc[nf]);
            }
        }
    }
    // ---- epilogue: bias + relu + feat_arm residual ----
#pragma unroll
    for (int nf = 0; nf < 4; nf++)
#pragma unroll
        for (int r = 0; r < 8; r++) {
            int m = m0 + r + 8 * h;
            int n = n0 + nf * 16 + lr;
            size_t idx = ((size_t)(b * C2 + m)) * Nn + n;
            float v = fmaxf(acc[nf][r] + dcnb[m], 0.f);
            out[idx] = v + feat_arm[idx];
        }
}

// ---------------- host launcher ----------------
extern "C" void kernel_launch(void* const* d_in, const int* in_sizes, int n_in,
                              void* d_out, int out_size, void* d_ws, size_t ws_size,
                              hipStream_t stream) {
    const float* feat_l = (const float*)d_in[0];
    const float* feat_s = (const float*)d_in[1];
    const float* w_at   = (const float*)d_in[2];
    const float* w_conv = (const float*)d_in[3];
    const float* w_offw = (const float*)d_in[4];
    const float* w_omw  = (const float*)d_in[5];
    const float* b_om   = (const float*)d_in[6];
    const float* w_dcn  = (const float*)d_in[7];
    const float* b_dcn  = (const float*)d_in[8];
    float* out = (float*)d_out;

    char* ws = (char*)d_ws;
    float* g     = (float*)(ws + 0);            //   2 KB
    float* scale = (float*)(ws + 2048);         //   2 KB
    u16* w_arm   = (u16*)(ws + 4096);           //  64 KB  [128][256]
    u16* w_off   = (u16*)(ws + 69632);          //  64 KB  [128][256]
    u16* w_om    = (u16*)(ws + 135168);         // 504 KB  [9][224][128]
    u16* w_dA    = (u16*)(ws + 651264);         // 320 KB  [8][128][160]
    u16* xl      = (u16*)(ws + 978944);         //  25 MB  [b][n][256]
    u16* xarm2   = (u16*)(ws + 27193344);       //  25 MB  [b][n][256]
    u16* offh    = (u16*)(ws + 53407744);       // 12.5 MB [b][n][128]
    u16* xsg     = (u16*)(ws + 66514944);       // 12.5 MB [b][d][n][16]
    float* farm  = (float*)(ws + 79622144);     //  25 MB  [b][128][n]
    float* om    = (float*)(ws + 105836544);    //  42 MB  [b][216][n]
    (void)in_sizes; (void)n_in; (void)out_size; (void)ws_size;

    int prep_total = 2 * C2 * C1 + 9 * MOMP * C2 + DGN * C2 * KDCN;   // 487424

    k_pool   <<<dim3(Bn * C1), 256, 0, stream>>>(feat_l, g);
    k_atten  <<<dim3(Bn),      256, 0, stream>>>(g, w_at, scale);
    k_prep_w <<<dim3((prep_total + 255) / 256), 256, 0, stream>>>(
        w_conv, w_offw, w_omw, w_dcn, w_arm, w_off, w_om, w_dA);
    k_trans_l<<<dim3(Nn / 64, C1 / 64, Bn), 256, 0, stream>>>(feat_l, scale, xl);
    k_trans_s<<<dim3(Nn / 64, C2 / 64, Bn), 256, 0, stream>>>(feat_s, xarm2, xsg);
    k_gemm1x1<0><<<dim3(Nn / 64, Bn), 256, 0, stream>>>(w_arm, xl, farm, xarm2);
    k_gemm1x1<1><<<dim3(Nn / 64, Bn), 256, 0, stream>>>(w_off, xarm2, nullptr, offh);
    k_conv_om<<<dim3(Hn, 2, Bn), 256, 0, stream>>>(w_om, offh, b_om, om);
    k_dcn    <<<dim3(Nn / 64, Bn), 256, 0, stream>>>(w_dA, xsg, om, b_dcn, farm, out);
}
